// GraphTripleConv_10668698763869
// MI455X (gfx1250) — matrix operand order repaired and checked by reading, not compiled, not run on hardware
//
#include <hip/hip_runtime.h>
#include <hip/hip_bf16.h>

typedef __attribute__((ext_vector_type(16))) _Float16 v16h;
typedef __attribute__((ext_vector_type(8)))  _Float16 v8h;
typedef __attribute__((ext_vector_type(8)))  float    v8f;

#define NUM_OBJS   50000
#define NUM_TRIP   200000
#define HDIM       512
#define DOUT       128
#define K1         384     // 2*128 + 128
#define N2         1152    // 2*512 + 128

// ---- tile config ----
#define BM 128
#define BN 128
#define BK 32
#define LDSA 40            // BK + 8 pad (halfwords) -> rows stay 16B aligned
#define LDSB 40

__device__ __forceinline__ v16h cat16(v8h lo, v8h hi) {
    return __builtin_shufflevector(lo, hi, 0,1,2,3,4,5,6,7,8,9,10,11,12,13,14,15);
}

// A fragment (16x32 f16): lane half kh: elems 0..7 -> K = kh*8..+7, elems 8..15 -> K = 16+kh*8..+7
__device__ __forceinline__ v16h frag_a(const _Float16* rowptr, int kh) {
    v8h lo = *(const v8h*)(rowptr + kh * 8);
    v8h hi = *(const v8h*)(rowptr + 16 + kh * 8);
    return cat16(lo, hi);
}
// B fragment (32x16 f16, staged transposed as Bs[n][k]): elems 0..7 -> K = kh*8..+7, elems 8..15 -> K = 16+kh*8..+7
__device__ __forceinline__ v16h frag_b(const _Float16* rowptr, int kh) {
    v8h lo = *(const v8h*)(rowptr + kh * 8);
    v8h hi = *(const v8h*)(rowptr + 16 + kh * 8);
    return cat16(lo, hi);
}

// Fused WMMA GEMM: C(MxN) = relu(A(MxK) @ B16(KxN) + bias)
// AMODE 0: A16 f16 row-major.  AMODE 1: gather-concat [obj[s]|pred|obj[o]] f32 -> f16.
// CMODE 0: f16 store. CMODE 1: scatter (pooled atomics + new_p). CMODE 2: f32 store.
template<int AMODE, int CMODE>
__global__ __launch_bounds__(256)
void gemm_wmma(const float* __restrict__ objv, const float* __restrict__ predv,
               const long long* __restrict__ edges,
               const _Float16* __restrict__ A16,
               const _Float16* __restrict__ B16, int ldb,
               const float* __restrict__ bias,
               int M, int N, int K,
               _Float16* __restrict__ Cf16,
               float* __restrict__ pooled, float* __restrict__ outP,
               float* __restrict__ Cf32) {
    __shared__ __align__(16) _Float16 As[BM * LDSA];
    __shared__ __align__(16) _Float16 Bs[BN * LDSB];
    __shared__ long long eStage[2 * BM];   // used only by CMODE==1

    const int tid  = threadIdx.x;
    const int lane = tid & 31;
    const int w    = tid >> 5;          // 8 waves
    const int wmBase = (w >> 1) * 32;   // 0,32,64,96
    const int wnBase = (w & 1) * 64;    // 0,64
    const int m0 = blockIdx.y * BM;
    const int n0 = blockIdx.x * BN;

    // ---- hoisted per-thread A staging state (loop-invariant) ----
    const int rA  = tid >> 1;           // 0..127: A-tile row this thread stages
    const int cbA = (tid & 1) * 16;     // 0 or 16
    const int growA = m0 + rA;
    const bool okA = growA < M;
    const float* rowS = nullptr;
    const float* rowP = nullptr;
    const float* rowO = nullptr;
    const _Float16* rowA16 = nullptr;
    if constexpr (AMODE == 1) {
        if (okA) {
            rowS = objv  + (size_t)edges[2 * growA]     * 128;
            rowP = predv + (size_t)growA                * 128;
            rowO = objv  + (size_t)edges[2 * growA + 1] * 128;
        }
    } else {
        if (okA) rowA16 = A16 + (size_t)growA * K;
    }
    // B staging coords
    const int krB = tid >> 3;           // 0..31
    const int nbB = (tid & 7) * 16;     // 0..112

    v8f acc[2][4];
#pragma unroll
    for (int mi = 0; mi < 2; ++mi)
#pragma unroll
        for (int ni = 0; ni < 4; ++ni) acc[mi][ni] = (v8f){};

    for (int k0 = 0; k0 < K; k0 += BK) {
        __syncthreads();
        // ---- stage A tile (BM x BK) ----
        if constexpr (AMODE == 0) {
            v8h d0 = {}, d1 = {};
            if (okA) {
                const _Float16* src = rowA16 + k0 + cbA;
                d0 = *(const v8h*)(src);
                d1 = *(const v8h*)(src + 8);
            }
            *(v8h*)(&As[rA * LDSA + cbA])     = d0;
            *(v8h*)(&As[rA * LDSA + cbA + 8]) = d1;
        } else {
            float buf[16];
#pragma unroll
            for (int j = 0; j < 16; ++j) buf[j] = 0.f;
            if (okA) {
                const int seg = k0 >> 7;               // 128-wide segments of K=384
                const int ks  = (k0 & 127) + cbA;
                const float* src = (seg == 0) ? rowS : ((seg == 1) ? rowP : rowO);
                src += ks;
#pragma unroll
                for (int j = 0; j < 4; ++j) {
                    float4 t = ((const float4*)src)[j];
                    buf[4 * j + 0] = t.x; buf[4 * j + 1] = t.y;
                    buf[4 * j + 2] = t.z; buf[4 * j + 3] = t.w;
                }
            }
            v8h d0, d1;
#pragma unroll
            for (int j = 0; j < 8; ++j) { d0[j] = (_Float16)buf[j]; d1[j] = (_Float16)buf[8 + j]; }
            *(v8h*)(&As[rA * LDSA + cbA])     = d0;
            *(v8h*)(&As[rA * LDSA + cbA + 8]) = d1;
        }
        // ---- stage B tile (BK x BN), transposed into Bs[n][k] ----
        {
            const _Float16* src = B16 + (size_t)(k0 + krB) * ldb + n0 + nbB;
            v8h bv0 = *(const v8h*)(src);
            v8h bv1 = *(const v8h*)(src + 8);
#pragma unroll
            for (int j = 0; j < 8; ++j) {
                Bs[(nbB + j)     * LDSB + krB] = bv0[j];
                Bs[(nbB + 8 + j) * LDSB + krB] = bv1[j];
            }
        }
        if (k0 + BK < K) __builtin_prefetch(B16 + (size_t)(k0 + BK) * ldb + n0, 0, 1);
        __syncthreads();

        // ---- compute: 2x4 WMMA frags per wave (32x64 tile) ----
        const int mloc = lane & 15;
        const int kh   = lane >> 4;
        v16h a[2];
        a[0] = frag_a(&As[(wmBase + mloc)      * LDSA], kh);
        a[1] = frag_a(&As[(wmBase + 16 + mloc) * LDSA], kh);
        v16h b[4];
#pragma unroll
        for (int ni = 0; ni < 4; ++ni)
            b[ni] = frag_b(&Bs[(wnBase + ni * 16 + mloc) * LDSB], kh);
#pragma unroll
        for (int mi = 0; mi < 2; ++mi)
#pragma unroll
            for (int ni = 0; ni < 4; ++ni)
                acc[mi][ni] = __builtin_amdgcn_wmma_f32_16x16x32_f16(
                    false, a[mi], false, b[ni], (short)0, acc[mi][ni], false, false);
    }

    // ---- epilogue ----
    if constexpr (CMODE == 1) {
        // stage this block's 128 edge pairs into LDS once
        const int row   = tid >> 1;
        const int which = tid & 1;
        const int gm = m0 + row;
        eStage[2 * row + which] = (gm < M) ? edges[2 * (size_t)gm + which] : 0;
        __syncthreads();
    }

    const int nloc = lane & 15;
    const int rsel = lane >> 4;
#pragma unroll
    for (int mi = 0; mi < 2; ++mi) {
#pragma unroll
        for (int ni = 0; ni < 4; ++ni) {
            const v8f c = acc[mi][ni];
            const int gn = n0 + wnBase + ni * 16 + nloc;
            const float bv = bias[gn];
#pragma unroll
            for (int r = 0; r < 8; ++r) {
                const int lm = wmBase + mi * 16 + r + 8 * rsel;
                const int gm = m0 + lm;
                if (gm >= M) continue;
                float v = c[r] + bv;
                v = v > 0.f ? v : 0.f;
                if constexpr (CMODE == 0) {
                    Cf16[(size_t)gm * N + gn] = (_Float16)v;
                } else if constexpr (CMODE == 1) {
                    if (gn < HDIM) {
                        atomicAdd(&pooled[(size_t)eStage[2 * lm] * HDIM + gn], v);
                    } else if (gn < HDIM + DOUT) {
                        outP[(size_t)gm * DOUT + (gn - HDIM)] = v;
                    } else {
                        atomicAdd(&pooled[(size_t)eStage[2 * lm + 1] * HDIM + (gn - HDIM - DOUT)], v);
                    }
                } else {
                    Cf32[(size_t)gm * N + gn] = v;
                }
            }
        }
    }
}

__global__ void cvt_f32_f16(const float* __restrict__ src, _Float16* __restrict__ dst, int n) {
    int i = blockIdx.x * blockDim.x + threadIdx.x;
    if (i < n) dst[i] = (_Float16)src[i];
}

__global__ void counts_kernel(const long long* __restrict__ edges, float* __restrict__ counts, int T) {
    int t = blockIdx.x * blockDim.x + threadIdx.x;
    if (t < T) {
        atomicAdd(&counts[edges[2 * t]],     1.0f);
        atomicAdd(&counts[edges[2 * t + 1]], 1.0f);
    }
}

__global__ void pool_div(const float* __restrict__ pooled, const float* __restrict__ counts,
                         _Float16* __restrict__ pooled16, int n) {
    int i = blockIdx.x * blockDim.x + threadIdx.x;
    if (i < n) {
        float c = fmaxf(counts[i >> 9], 1.0f);   // HDIM == 512
        pooled16[i] = (_Float16)(pooled[i] / c);
    }
}

extern "C" void kernel_launch(void* const* d_in, const int* in_sizes, int n_in,
                              void* d_out, int out_size, void* d_ws, size_t ws_size,
                              hipStream_t stream) {
    (void)in_sizes; (void)n_in; (void)out_size; (void)ws_size;
    const float*     obj   = (const float*)d_in[0];
    const float*     pred  = (const float*)d_in[1];
    const long long* edges = (const long long*)d_in[2];
    const float* W1a = (const float*)d_in[3];
    const float* b1a = (const float*)d_in[4];
    const float* W1b = (const float*)d_in[5];
    const float* b1b = (const float*)d_in[6];
    const float* W2a = (const float*)d_in[7];
    const float* b2a = (const float*)d_in[8];
    const float* W2b = (const float*)d_in[9];
    const float* b2b = (const float*)d_in[10];

    float* out_obj = (float*)d_out;                              // 50000 x 128
    float* out_p   = (float*)d_out + (size_t)NUM_OBJS * DOUT;    // 200000 x 128

    // ---- carve workspace (aligned 256B) ----
    char* ws = (char*)d_ws;
    size_t off = 0;
    auto carve = [&](size_t bytes) {
        void* p = ws + off;
        off = (off + bytes + 255) & ~(size_t)255;
        return p;
    };
    _Float16* hA       = (_Float16*)carve((size_t)NUM_TRIP * HDIM * 2);
    float*    pooled   = (float*)   carve((size_t)NUM_OBJS * HDIM * 4);
    float*    counts   = (float*)   carve((size_t)NUM_OBJS * 4);
    _Float16* pooled16 = (_Float16*)carve((size_t)NUM_OBJS * HDIM * 2);
    _Float16* h2       = (_Float16*)carve((size_t)NUM_OBJS * HDIM * 2);
    _Float16* W1a16    = (_Float16*)carve((size_t)K1 * HDIM * 2);
    _Float16* W1b16    = (_Float16*)carve((size_t)HDIM * N2 * 2);
    _Float16* W2a16    = (_Float16*)carve((size_t)HDIM * HDIM * 2);
    _Float16* W2b16    = (_Float16*)carve((size_t)HDIM * DOUT * 2);

    hipMemsetAsync(pooled, 0, (size_t)NUM_OBJS * HDIM * 4, stream);
    hipMemsetAsync(counts, 0, (size_t)NUM_OBJS * 4, stream);

    cvt_f32_f16<<<(K1 * HDIM + 255) / 256, 256, 0, stream>>>(W1a, W1a16, K1 * HDIM);
    cvt_f32_f16<<<(HDIM * N2 + 255) / 256, 256, 0, stream>>>(W1b, W1b16, HDIM * N2);
    cvt_f32_f16<<<(HDIM * HDIM + 255) / 256, 256, 0, stream>>>(W2a, W2a16, HDIM * HDIM);
    cvt_f32_f16<<<(HDIM * DOUT + 255) / 256, 256, 0, stream>>>(W2b, W2b16, HDIM * DOUT);

    counts_kernel<<<(NUM_TRIP + 255) / 256, 256, 0, stream>>>(edges, counts, NUM_TRIP);

    const int mt1 = (NUM_TRIP + BM - 1) / BM;   // 1563
    const int mt2 = (NUM_OBJS + BM - 1) / BM;   // 391

    // GEMM1: gather-concat (200000 x 384) @ W1a -> hA f16
    gemm_wmma<1, 0><<<dim3(HDIM / BN, mt1), 256, 0, stream>>>(
        obj, pred, edges, nullptr, W1a16, HDIM, b1a,
        NUM_TRIP, HDIM, K1, hA, nullptr, nullptr, nullptr);

    // GEMM2: hA @ W1b -> scatter (pooled atomics + new_p)
    gemm_wmma<0, 1><<<dim3(N2 / BN, mt1), 256, 0, stream>>>(
        nullptr, nullptr, edges, hA, W1b16, N2, b1b,
        NUM_TRIP, N2, HDIM, nullptr, pooled, out_p, nullptr);

    // pooled /= max(counts,1), -> f16
    pool_div<<<((NUM_OBJS * HDIM) + 255) / 256, 256, 0, stream>>>(
        pooled, counts, pooled16, NUM_OBJS * HDIM);

    // GEMM3: pooled16 @ W2a -> h2 f16
    gemm_wmma<0, 0><<<dim3(HDIM / BN, mt2), 256, 0, stream>>>(
        nullptr, nullptr, edges, pooled16, W2a16, HDIM, b2a,
        NUM_OBJS, HDIM, HDIM, h2, nullptr, nullptr, nullptr);

    // GEMM4: h2 @ W2b -> new_obj f32 (d_out)
    gemm_wmma<0, 2><<<dim3(DOUT / BN, mt2), 256, 0, stream>>>(
        nullptr, nullptr, edges, h2, W2b16, DOUT, b2b,
        NUM_OBJS, DOUT, HDIM, nullptr, nullptr, nullptr, out_obj);
}
